// MultiHeadAttention_3710851744198
// MI455X (gfx1250) — matrix operand order repaired and checked by reading, not compiled, not run on hardware
//
#include <hip/hip_runtime.h>
#include <hip/hip_bf16.h>

// ---------------------------------------------------------------------------
// MI455X (gfx1250) Multi-Head Attention forward, bf16 WMMA + f32 accumulate.
// B=2, S=2048, H=16, Dk=64, D_MODEL=1024.
// Pipeline: cvt(x) / cvt+transpose(W) -> QKV GEMM -> RoPE/split -> flash attn
//           -> output GEMM. GEMMs use double-buffered LDS with async
//           global->LDS copies (ASYNCcnt) when available.
// ---------------------------------------------------------------------------

typedef __bf16 bf16_t;
typedef __attribute__((ext_vector_type(16))) __bf16 v16bf;
typedef __attribute__((ext_vector_type(8)))  float  v8f;
typedef __attribute__((ext_vector_type(4)))  unsigned int u32x4;

#define BDIM 2
#define SDIM 2048
#define HDIM 16
#define DKDIM 64
#define DMODEL 1024

union FragB16 { v16bf v; u32x4 q[2]; };

static __device__ inline v8f v8f_zero() {
  v8f z = {0.f, 0.f, 0.f, 0.f, 0.f, 0.f, 0.f, 0.f};
  return z;
}

#if defined(__gfx1250__) && __has_builtin(__builtin_amdgcn_global_load_async_to_lds_b128) && \
    __has_builtin(__builtin_amdgcn_s_wait_asynccnt)
#define ASYNC_COPY 1
// The builtin's parameter type (from the compiler diagnostic) is a pointer to
// a GCC-style int4 vector: AS(1) for the global source, AS(3) for LDS dest.
typedef int i32x4v __attribute__((vector_size(16)));
typedef __attribute__((address_space(1))) i32x4v GV4;
typedef __attribute__((address_space(3))) i32x4v LV4;
static __device__ inline GV4* as_g(const void* p) { return (GV4*)p; }
static __device__ inline LV4* as_l(void* p)       { return (LV4*)p; }
#else
#define ASYNC_COPY 0
#endif

// ---------------------------------------------------------------------------
// f32 -> bf16 conversion (x)
// ---------------------------------------------------------------------------
__global__ void cvt_f32_bf16(const float* __restrict__ in,
                             bf16_t* __restrict__ out, int n) {
  int i = blockIdx.x * blockDim.x + threadIdx.x;
  if (i < n) out[i] = (bf16_t)in[i];
}

// ---------------------------------------------------------------------------
// f32 [K,N] -> bf16 [N,K] transpose (weights). Read coalesced, write strided;
// one-time cost so the GEMM hot loop gets pure b128 LDS tile copies.
// ---------------------------------------------------------------------------
__global__ void cvt_transpose_f32_bf16(const float* __restrict__ in,  // [K,N]
                                       bf16_t* __restrict__ out,      // [N,K]
                                       int K, int N) {
  int i = blockIdx.x * blockDim.x + threadIdx.x;
  if (i < K * N) {
    int k = i / N, n = i - k * N;
    out[(size_t)n * K + k] = (bf16_t)in[i];
  }
}

// ---------------------------------------------------------------------------
// C[M,N] (f32) = A[M,K] (bf16 row-major) * Bt[N,K]^T (bf16, pre-transposed)
// Block tile 128x128x32. 256 threads = 8 waves; wave grid 2(M) x 4(N);
// each wave computes 64x32 = 4x2 WMMA accumulators. Double-buffered LDS;
// async global->LDS copies overlap with the WMMAs when available.
// ---------------------------------------------------------------------------
__launch_bounds__(256)
__global__ void gemm_bf16_wmma(const bf16_t* __restrict__ A,
                               const bf16_t* __restrict__ Bt,
                               float* __restrict__ C,
                               int M, int N, int K) {
  __shared__ __align__(16) bf16_t sA[2][128 * 40];   // [row][k], stride 40
  __shared__ __align__(16) bf16_t sB[2][128 * 40];   // [n][k],  stride 40

  const int tid  = threadIdx.x;
  const int lane = tid & 31;
  const int wid  = tid >> 5;
  const int wm   = wid & 1;
  const int wn   = wid >> 1;
  const int l16  = lane & 15;
  const int lhi  = lane >> 4;
  const int blockM = blockIdx.y * 128;
  const int blockN = blockIdx.x * 128;

  const bf16_t* Ablk = A  + (size_t)blockM * K;
  const bf16_t* Bblk = Bt + (size_t)blockN * K;

  // This thread's two 16B segments of each 128x32 tile (512 segments total).
  const int r0 = tid >> 2,          c0 = (tid & 3) * 8;
  const int r1 = (tid + 256) >> 2,  c1 = ((tid + 256) & 3) * 8;

  v8f acc[4][2];
#pragma unroll
  for (int mt = 0; mt < 4; ++mt)
#pragma unroll
    for (int nt = 0; nt < 2; ++nt) acc[mt][nt] = v8f_zero();

  auto copy_tiles = [&](int buf, int kt) {
#if ASYNC_COPY
    __builtin_amdgcn_global_load_async_to_lds_b128(
        as_g(Ablk + (size_t)r0 * K + kt + c0), as_l(&sA[buf][r0 * 40 + c0]), 0, 0);
    __builtin_amdgcn_global_load_async_to_lds_b128(
        as_g(Ablk + (size_t)r1 * K + kt + c1), as_l(&sA[buf][r1 * 40 + c1]), 0, 0);
    __builtin_amdgcn_global_load_async_to_lds_b128(
        as_g(Bblk + (size_t)r0 * K + kt + c0), as_l(&sB[buf][r0 * 40 + c0]), 0, 0);
    __builtin_amdgcn_global_load_async_to_lds_b128(
        as_g(Bblk + (size_t)r1 * K + kt + c1), as_l(&sB[buf][r1 * 40 + c1]), 0, 0);
#else
    u32x4 a0 = *(const u32x4*)(Ablk + (size_t)r0 * K + kt + c0);
    u32x4 a1 = *(const u32x4*)(Ablk + (size_t)r1 * K + kt + c1);
    u32x4 b0 = *(const u32x4*)(Bblk + (size_t)r0 * K + kt + c0);
    u32x4 b1 = *(const u32x4*)(Bblk + (size_t)r1 * K + kt + c1);
    *(u32x4*)(&sA[buf][r0 * 40 + c0]) = a0;
    *(u32x4*)(&sA[buf][r1 * 40 + c1]) = a1;
    *(u32x4*)(&sB[buf][r0 * 40 + c0]) = b0;
    *(u32x4*)(&sB[buf][r1 * 40 + c1]) = b1;
#endif
  };

  auto sync_tiles = [&]() {
#if ASYNC_COPY
    __builtin_amdgcn_s_wait_asynccnt(0);
#endif
    __syncthreads();
  };

  copy_tiles(0, 0);
  sync_tiles();

  for (int kt = 0; kt < K; kt += 32) {
    const int cur = (kt >> 5) & 1;
    if (kt + 32 < K) copy_tiles(cur ^ 1, kt + 32);   // overlap with WMMAs

    FragB16 af[4], bfr[2];
#pragma unroll
    for (int mt = 0; mt < 4; ++mt) {
      int m  = wm * 64 + mt * 16 + l16;
      int kb = lhi * 8;
      af[mt].q[0] = *(const u32x4*)(&sA[cur][m * 40 + kb]);       // K kb..kb+7
      af[mt].q[1] = *(const u32x4*)(&sA[cur][m * 40 + kb + 16]);  // K kb+16..+23
    }
#pragma unroll
    for (int nt = 0; nt < 2; ++nt) {
      int n  = wn * 32 + nt * 16 + l16;
      int kb = lhi * 8;
      bfr[nt].q[0] = *(const u32x4*)(&sB[cur][n * 40 + kb]);      // K kb..kb+7
      bfr[nt].q[1] = *(const u32x4*)(&sB[cur][n * 40 + kb + 16]); // K kb+16..+23
    }
#pragma unroll
    for (int mt = 0; mt < 4; ++mt)
#pragma unroll
      for (int nt = 0; nt < 2; ++nt)
        acc[mt][nt] = __builtin_amdgcn_wmma_f32_16x16x32_bf16(
            false, af[mt].v, false, bfr[nt].v, (short)0, acc[mt][nt], false, false);

    sync_tiles();
  }

  // Writeback: lane holds column n, rows r + 8*lhi (C layout); incremental
  // row pointer avoids per-store 64-bit multiplies.
#pragma unroll
  for (int mt = 0; mt < 4; ++mt) {
#pragma unroll
    for (int nt = 0; nt < 2; ++nt) {
      float* cp = C + (size_t)(blockM + wm * 64 + mt * 16 + 8 * lhi) * N
                    + (blockN + wn * 32 + nt * 16 + l16);
#pragma unroll
      for (int r = 0; r < 8; ++r) {
        *cp = acc[mt][nt][r];
        cp += N;
      }
    }
  }
}

// ---------------------------------------------------------------------------
// RoPE + head split: qkv f32 [B*S, 3*D] -> Q,K bf16 [B,H,S,64] (Q pre-scaled
// by 0.125, exact) and V transposed bf16 [B,H,64,S].
// ---------------------------------------------------------------------------
__global__ void rope_split(const float* __restrict__ qkv,
                           bf16_t* __restrict__ Q,
                           bf16_t* __restrict__ Kd,
                           bf16_t* __restrict__ Vt) {
  int i = blockIdx.x * blockDim.x + threadIdx.x;     // 2^21 threads
  int j = i & 31;
  int h = (i >> 5) & 15;
  int s = (i >> 9) & 2047;
  int b = i >> 20;

  const float* row = qkv + (size_t)(b * SDIM + s) * (3 * DMODEL);
  float q1 = row[h * 64 + j],           q2 = row[h * 64 + j + 32];
  float k1 = row[DMODEL + h * 64 + j],  k2 = row[DMODEL + h * 64 + j + 32];
  float v1 = row[2 * DMODEL + h * 64 + j];
  float v2 = row[2 * DMODEL + h * 64 + j + 32];

  float inv = exp2f(-(float)j * (13.287712379549449f / 32.0f));  // 10000^(-j/32)
  float ang = (float)s * inv;
  float sn, cs;
  __sincosf(ang, &sn, &cs);

  float qo1 = ( q1 * cs + q2 * sn) * 0.125f;
  float qo2 = (-q1 * sn + q2 * cs) * 0.125f;
  float ko1 =   k1 * cs + k2 * sn;
  float ko2 =  -k1 * sn + k2 * cs;

  size_t head = (size_t)b * HDIM + h;
  bf16_t* qp = Q  + (head * SDIM + s) * DKDIM;
  bf16_t* kp = Kd + (head * SDIM + s) * DKDIM;
  qp[j] = (bf16_t)qo1; qp[j + 32] = (bf16_t)qo2;
  kp[j] = (bf16_t)ko1; kp[j + 32] = (bf16_t)ko2;
  bf16_t* vp = Vt + head * (size_t)DKDIM * SDIM;
  vp[(size_t)j * SDIM + s]        = (bf16_t)v1;
  vp[(size_t)(j + 32) * SDIM + s] = (bf16_t)v2;
}

// ---------------------------------------------------------------------------
// Causal flash attention. 256 threads = 8 independent waves; each wave owns
// 16 queries. Key blocks of 64: 8 WMMAs for QK^T + 8 WMMAs for P*V per block.
// Per-wave LDS tile transposes P (C-layout -> A-layout); same-wave LDS ops
// are in-order so no workgroup barrier is needed. All WMMA control flow is
// wave-uniform (EXEC all ones).
// ---------------------------------------------------------------------------
__launch_bounds__(256)
__global__ void flash_attn_wmma(const bf16_t* __restrict__ Q,
                                const bf16_t* __restrict__ Kd,
                                const bf16_t* __restrict__ Vt,
                                bf16_t* __restrict__ Out) {
  __shared__ __align__(16) bf16_t sP[8][16 * 72];   // per-wave 16x64 P tile

  const int lane = threadIdx.x & 31;
  const int wid  = threadIdx.x >> 5;
  const int l16  = lane & 15;
  const int lhi  = lane >> 4;

  int blk  = blockIdx.x;                 // B*H*(S/128) blocks
  int qb   = blk & (SDIM / 128 - 1);
  int head = blk >> 4;                   // b*H + h
  int q0   = qb * 128 + wid * 16;

  const bf16_t* qh = Q  + (size_t)head * SDIM * DKDIM;
  const bf16_t* kh = Kd + (size_t)head * SDIM * DKDIM;
  const bf16_t* vh = Vt + (size_t)head * DKDIM * SDIM;

  FragB16 qf[2];
#pragma unroll
  for (int ks = 0; ks < 2; ++ks) {
    int m  = q0 + l16;
    int kb = ks * 32 + lhi * 8;
    qf[ks].q[0] = *(const u32x4*)(qh + (size_t)m * DKDIM + kb);
    qf[ks].q[1] = *(const u32x4*)(qh + (size_t)m * DKDIM + kb + 16);
  }

  v8f oacc[4];
#pragma unroll
  for (int nt = 0; nt < 4; ++nt) oacc[nt] = v8f_zero();
  float rmax[8], rsum[8];
#pragma unroll
  for (int r = 0; r < 8; ++r) { rmax[r] = -1e30f; rsum[r] = 0.0f; }

  bf16_t* myP = sP[wid];
  int nkb = (q0 + 15) / 64 + 1;

  for (int kbi = 0; kbi < nkb; ++kbi) {
    int k0 = kbi * 64;

    // Prefetch next key block's K rows and V columns (wave-uniform branch).
    if (kbi + 1 < nkb) {
      const bf16_t* nk = kh + (size_t)(k0 + 64 + lane) * DKDIM;
      __builtin_prefetch(nk, 0, 1);
      __builtin_prefetch(nk + 32 * DKDIM, 0, 1);
      const bf16_t* nv = vh + (size_t)lane * SDIM + (k0 + 64);
      __builtin_prefetch(nv, 0, 1);
      __builtin_prefetch(nv + (size_t)32 * SDIM, 0, 1);
    }

    // ---- S = Q * K^T (Q pre-scaled) ----
    v8f st[4];
#pragma unroll
    for (int nt = 0; nt < 4; ++nt) st[nt] = v8f_zero();
#pragma unroll
    for (int ks = 0; ks < 2; ++ks) {
#pragma unroll
      for (int nt = 0; nt < 4; ++nt) {
        FragB16 bfr;
        int key = k0 + nt * 16 + l16;
        int kk  = ks * 32 + lhi * 8;
        bfr.q[0] = *(const u32x4*)(kh + (size_t)key * DKDIM + kk);
        bfr.q[1] = *(const u32x4*)(kh + (size_t)key * DKDIM + kk + 16);
        st[nt] = __builtin_amdgcn_wmma_f32_16x16x32_bf16(
            false, qf[ks].v, false, bfr.v, (short)0, st[nt], false, false);
      }
    }

    // ---- causal mask (per-lane selects; branch is wave-uniform) ----
    if (k0 + 63 > q0) {
#pragma unroll
      for (int nt = 0; nt < 4; ++nt) {
        int key = k0 + nt * 16 + l16;
#pragma unroll
        for (int r = 0; r < 8; ++r) {
          int m = q0 + r + 8 * lhi;
          st[nt][r] = (key > m) ? -1e30f : st[nt][r];
        }
      }
    }

    // ---- online softmax + P -> LDS (bf16, A-layout transpose) ----
#pragma unroll
    for (int r = 0; r < 8; ++r) {
      float mx = fmaxf(fmaxf(st[0][r], st[1][r]), fmaxf(st[2][r], st[3][r]));
#pragma unroll
      for (int off = 8; off; off >>= 1) mx = fmaxf(mx, __shfl_xor(mx, off, 32));
      float nm   = fmaxf(rmax[r], mx);
      float corr = __expf(rmax[r] - nm);
      rmax[r] = nm;
      float ls = 0.f;
#pragma unroll
      for (int nt = 0; nt < 4; ++nt) {
        float p = __expf(st[nt][r] - nm);
        st[nt][r] = p;
        ls += p;
      }
#pragma unroll
      for (int off = 8; off; off >>= 1) ls += __shfl_xor(ls, off, 32);
      rsum[r] = rsum[r] * corr + ls;
      oacc[0][r] *= corr; oacc[1][r] *= corr;
      oacc[2][r] *= corr; oacc[3][r] *= corr;
      int mloc = r + 8 * lhi;
#pragma unroll
      for (int nt = 0; nt < 4; ++nt)
        myP[mloc * 72 + nt * 16 + l16] = (bf16_t)st[nt][r];
    }

    // ---- O += P * V (same-wave LDS in-order: no barrier needed) ----
#pragma unroll
    for (int ks = 0; ks < 2; ++ks) {
      FragB16 pf;
      int kb = ks * 32 + lhi * 8;
      pf.q[0] = *(const u32x4*)(myP + l16 * 72 + kb);
      pf.q[1] = *(const u32x4*)(myP + l16 * 72 + kb + 16);
#pragma unroll
      for (int nt = 0; nt < 4; ++nt) {
        FragB16 vf;
        int d  = nt * 16 + l16;
        int kk = k0 + ks * 32 + lhi * 8;
        vf.q[0] = *(const u32x4*)(vh + (size_t)d * SDIM + kk);
        vf.q[1] = *(const u32x4*)(vh + (size_t)d * SDIM + kk + 16);
        oacc[nt] = __builtin_amdgcn_wmma_f32_16x16x32_bf16(
            false, pf.v, false, vf.v, (short)0, oacc[nt], false, false);
      }
    }
  }

  // ---- epilogue: normalize and store bf16 [B,S,D_MODEL] ----
  int b = head >> 4;
  int h = head & 15;
#pragma unroll
  for (int r = 0; r < 8; ++r) {
    float inv = 1.0f / rsum[r];
    int s = q0 + r + 8 * lhi;
    size_t rowoff = (size_t)(b * SDIM + s) * DMODEL + h * DKDIM;
#pragma unroll
    for (int nt = 0; nt < 4; ++nt)
      Out[rowoff + nt * 16 + l16] = (bf16_t)(oacc[nt][r] * inv);
  }
}

// ---------------------------------------------------------------------------
// Launch pipeline.
// ---------------------------------------------------------------------------
extern "C" void kernel_launch(void* const* d_in, const int* in_sizes, int n_in,
                              void* d_out, int out_size, void* d_ws, size_t ws_size,
                              hipStream_t stream) {
  (void)in_sizes; (void)n_in; (void)out_size; (void)ws_size;
  const float* x     = (const float*)d_in[0];
  const float* w_qkv = (const float*)d_in[1];
  const float* w_o   = (const float*)d_in[2];
  float* out = (float*)d_out;
  char*  ws  = (char*)d_ws;

  const size_t MB = 1024ull * 1024ull;
  bf16_t* xb    = (bf16_t*)(ws + 0 * MB);    // 8 MB  : x bf16        [4096,1024]
  bf16_t* wqkvT = (bf16_t*)(ws + 8 * MB);    // 6 MB  : w_qkv^T bf16  [3072,1024]
  bf16_t* woT   = (bf16_t*)(ws + 14 * MB);   // 2 MB  : w_o^T bf16    [1024,1024]
  float*  qkv   = (float*)(ws + 16 * MB);    // 48 MB : qkv f32       [4096,3072]
  bf16_t* Qr    = (bf16_t*)(ws + 64 * MB);   // 8 MB  : Q bf16        [B,H,S,64]
  bf16_t* Kr    = (bf16_t*)(ws + 72 * MB);   // 8 MB  : K bf16        [B,H,S,64]
  bf16_t* Vt    = (bf16_t*)(ws + 80 * MB);   // 8 MB  : V^T bf16      [B,H,64,S]
  bf16_t* attn  = (bf16_t*)(ws + 88 * MB);   // 8 MB  : attn out bf16 [4096,1024]

  const int M = BDIM * SDIM;                 // 4096

  int nx = M * DMODEL;
  cvt_f32_bf16<<<(nx + 255) / 256, 256, 0, stream>>>(x, xb, nx);
  int nw1 = DMODEL * 3 * DMODEL;
  cvt_transpose_f32_bf16<<<(nw1 + 255) / 256, 256, 0, stream>>>(
      w_qkv, wqkvT, DMODEL, 3 * DMODEL);
  int nw2 = DMODEL * DMODEL;
  cvt_transpose_f32_bf16<<<(nw2 + 255) / 256, 256, 0, stream>>>(
      w_o, woT, DMODEL, DMODEL);

  // QKV projection: [4096,1024] x [1024,3072]
  gemm_bf16_wmma<<<dim3(3 * DMODEL / 128, M / 128), 256, 0, stream>>>(
      xb, wqkvT, qkv, M, 3 * DMODEL, DMODEL);

  int nr = BDIM * SDIM * HDIM * 32;
  rope_split<<<nr / 256, 256, 0, stream>>>(qkv, Qr, Kr, Vt);

  flash_attn_wmma<<<BDIM * HDIM * (SDIM / 128), 256, 0, stream>>>(Qr, Kr, Vt, attn);

  // Output projection: [4096,1024] x [1024,1024] -> f32 d_out
  gemm_bf16_wmma<<<dim3(DMODEL / 128, M / 128), 256, 0, stream>>>(
      attn, woT, out, M, DMODEL, DMODEL);
}
